// AttentionModule_65068754535057
// MI455X (gfx1250) — matrix operand order repaired and checked by reading, not compiled, not run on hardware
//
#include <hip/hip_runtime.h>

#define H_ 8
#define D_ 256
#define E_ 1536
#define F_ 768
#define B_ 64
#define L_ 256
#define NEG_SLOPE 0.4f
#define MASK_FILL -100000000.0f

typedef __bf16 bf16_t;
typedef __attribute__((ext_vector_type(16))) __bf16 v16bf;
typedef __attribute__((ext_vector_type(8)))  __bf16 v8bf;
typedef __attribute__((ext_vector_type(8)))  float  v8f;

// Native CDNA5 V_TANH_F32 (single TRANS op) when available; libm fallback.
#if defined(__has_builtin)
#if __has_builtin(__builtin_amdgcn_tanhf)
#define FAST_TANH(x) __builtin_amdgcn_tanhf(x)
#endif
#endif
#ifndef FAST_TANH
#define FAST_TANH(x) tanhf(x)
#endif

// ---------------------------------------------------------------------------
// fp32 -> bf16 conversion (grid-stride)
// ---------------------------------------------------------------------------
__global__ void cvt_bf16_kernel(const float* __restrict__ src,
                                bf16_t* __restrict__ dst, int n) {
    int i = blockIdx.x * blockDim.x + threadIdx.x;
    int stride = gridDim.x * blockDim.x;
    for (; i < n; i += stride) dst[i] = (bf16_t)src[i];
}

// ---------------------------------------------------------------------------
// A-fragment (16x32 bf16, ISA 7.12.2): lane holds row (lane&15); its 16
// elements are K = [abase..abase+7] and [16+abase..23+abase].
// `p` already includes the +abase offset; second chunk sits at p+16.
// ---------------------------------------------------------------------------
__device__ __forceinline__ v16bf load_a16x32(const bf16_t* p) {
    v8bf lo = *(const v8bf*)(p);
    v8bf hi = *(const v8bf*)(p + 16);
    v16bf a;
#pragma unroll
    for (int i = 0; i < 8; ++i) { a[i] = lo[i]; a[i + 8] = hi[i]; }
    return a;
}

// ---------------------------------------------------------------------------
// q[h,b,d] = tanh(Query[h] @ x^T): per head a 64x256 GEMM, K = 1536.
// One block per head, 8 waves, each wave owns 8 of the 64 16x16 C tiles.
// ---------------------------------------------------------------------------
__launch_bounds__(256)
__global__ void q_gemm_kernel(const bf16_t* __restrict__ xbf,
                              const bf16_t* __restrict__ querybf,
                              float* __restrict__ qv) {
    const int h    = blockIdx.x;
    const int tid  = threadIdx.x;
    const int wave = tid >> 5;
    const int lane = tid & 31;
    const int laneHalf = lane >> 4;   // 0: lanes 0-15, 1: lanes 16-31
    const int l16      = lane & 15;
    const int abase = laneHalf * 8;   // A fragment K sub-offset
    const int kbase = laneHalf * 16;  // unused; B fragment also uses abase

#pragma unroll 1
    for (int i = 0; i < 8; ++i) {
        const int t  = wave * 8 + i;
        const int b0 = (t >> 4) * 16;   // M tile over B (4 tiles)
        const int d0 = (t & 15) * 16;   // N tile over D (16 tiles)
        const bf16_t* arow = xbf + (size_t)(b0 + l16) * E_;
        const bf16_t* brow = querybf + ((size_t)h * D_ + d0 + l16) * E_;

        v8f acc = {};
#pragma unroll 4
        for (int kk = 0; kk < E_; kk += 32) {
            v16bf a  = load_a16x32(arow + kk + abase);
            v16bf bb = load_a16x32(brow + kk + abase);
            acc = __builtin_amdgcn_wmma_f32_16x16x32_bf16(
                false, a, false, bb, (short)0, acc, false, false);
        }
        // C layout: lanes 0-15 hold M=r, lanes 16-31 hold M=r+8; N = l16
#pragma unroll
        for (int r = 0; r < 8; ++r) {
            const int m = r + laneHalf * 8;
            qv[((size_t)h * B_ + b0 + m) * D_ + d0 + l16] = FAST_TANH(acc[r]);
        }
    }
}

// ---------------------------------------------------------------------------
// Fused per-(h,b) attention:
//   k = tanh(bank_b @ Key_h^T)  (256x256, K=768, bf16 WMMA, lives in VGPRs)
//   score[l] = sum_d k[l,d] * q[h,b,d]   (folded into the C fragments)
//   masked softmax over L, then emb = attn @ bank (fp32 GEMV), LeakyReLU.
// Loop order: K outside d-tiles, 8 live accumulators -> each A (bank)
// fragment is loaded once per 8 WMMAs instead of once per WMMA.
// ---------------------------------------------------------------------------
__launch_bounds__(256)
__global__ void attn_fused_kernel(const bf16_t* __restrict__ bankbf,
                                  const bf16_t* __restrict__ keybf,
                                  const float* __restrict__ qv,
                                  const float* __restrict__ bank_f32,
                                  const int*  __restrict__ mask,
                                  float* __restrict__ out) {
    const int h = blockIdx.x / B_;
    const int b = blockIdx.x % B_;
    const int tid  = threadIdx.x;
    const int wave = tid >> 5;
    const int lane = tid & 31;
    const int laneHalf = lane >> 4;
    const int l16      = lane & 15;
    const int abase = laneHalf * 8;
    const int kbase = laneHalf * 16;

    __shared__ float q_s[D_];
    __shared__ float sc_s[L_];
    __shared__ float red_s[8];

    for (int i = tid; i < D_; i += 256)
        q_s[i] = qv[((size_t)h * B_ + b) * D_ + i];
    __syncthreads();

    const bf16_t* bankb = bankbf + (size_t)b * L_ * F_;
    const bf16_t* keyh  = keybf  + (size_t)h * D_ * F_;

    // Each wave covers two 16-row l-tiles (8 waves * 2 * 16 = 256 = L).
#pragma unroll 1
    for (int lt = 0; lt < 2; ++lt) {
        const int l0 = (wave * 2 + lt) * 16;
        const bf16_t* arow = bankb + (size_t)(l0 + l16) * F_ + abase;

        float part[8];
#pragma unroll
        for (int r = 0; r < 8; ++r) part[r] = 0.0f;

        // D split in two halves of 8 d-tiles to bound accumulator VGPRs
        // at 64 (8 x v8f) and avoid spills.
#pragma unroll 1
        for (int dh = 0; dh < 2; ++dh) {
            v8f acc[8];
#pragma unroll
            for (int j = 0; j < 8; ++j) acc[j] = (v8f){};

#pragma unroll 1
            for (int kk = 0; kk < F_; kk += 32) {
                v16bf a = load_a16x32(arow + kk);
                if (kk + 32 < F_)
                    __builtin_prefetch(arow + kk + 32, 0, 0);
#pragma unroll
                for (int j = 0; j < 8; ++j) {
                    const int d0 = (dh * 8 + j) * 16;
                    v16bf bb = load_a16x32(
                        keyh + (size_t)(d0 + l16) * F_ + kk + abase);
                    acc[j] = __builtin_amdgcn_wmma_f32_16x16x32_bf16(
                        false, a, false, bb, (short)0, acc[j], false, false);
                }
            }
            // tanh on f32 fragments (native v_tanh_f32), dot with q along
            // this lane's column d = d0 + l16.
#pragma unroll
            for (int j = 0; j < 8; ++j) {
                const int d0 = (dh * 8 + j) * 16;
                const float qd = q_s[d0 + l16];
#pragma unroll
                for (int r = 0; r < 8; ++r)
                    part[r] += FAST_TANH(acc[j][r]) * qd;
            }
        }
        // Reduce row partials across the 16 lanes of each half (wave32)
#pragma unroll
        for (int r = 0; r < 8; ++r) {
            float v = part[r];
            v += __shfl_xor(v, 1, 32);
            v += __shfl_xor(v, 2, 32);
            v += __shfl_xor(v, 4, 32);
            v += __shfl_xor(v, 8, 32);
            part[r] = v;
        }
        if (l16 == 0) {
#pragma unroll
            for (int r = 0; r < 8; ++r)
                sc_s[l0 + laneHalf * 8 + r] = part[r];
        }
    }
    __syncthreads();

    // ---- masked softmax over L = 256 (one thread per l) ----
    float s = sc_s[tid];
    if (mask[(size_t)b * L_ + tid] == 0) s = MASK_FILL;

    float m = s;
    m = fmaxf(m, __shfl_xor(m, 16, 32));
    m = fmaxf(m, __shfl_xor(m,  8, 32));
    m = fmaxf(m, __shfl_xor(m,  4, 32));
    m = fmaxf(m, __shfl_xor(m,  2, 32));
    m = fmaxf(m, __shfl_xor(m,  1, 32));
    if (lane == 0) red_s[wave] = m;
    __syncthreads();
    float mx = red_s[0];
#pragma unroll
    for (int i = 1; i < 8; ++i) mx = fmaxf(mx, red_s[i]);

    const float e = __expf(s - mx);
    float ssum = e;
    ssum += __shfl_xor(ssum, 16, 32);
    ssum += __shfl_xor(ssum,  8, 32);
    ssum += __shfl_xor(ssum,  4, 32);
    ssum += __shfl_xor(ssum,  2, 32);
    ssum += __shfl_xor(ssum,  1, 32);
    __syncthreads();                 // done reading red_s (max)
    if (lane == 0) red_s[wave] = ssum;
    __syncthreads();
    float tot = red_s[0];
#pragma unroll
    for (int i = 1; i < 8; ++i) tot += red_s[i];

    sc_s[tid] = e / tot;             // attn weights
    __syncthreads();

    // ---- emb[b,h,f] = LeakyReLU(sum_l attn[l] * bank[b,l,f]), fp32 ----
    const float* bankrow = bank_f32 + (size_t)b * L_ * F_;
#pragma unroll 1
    for (int f = tid; f < F_; f += 256) {
        float acc = 0.0f;
#pragma unroll 4
        for (int l = 0; l < L_; ++l)
            acc = fmaf(sc_s[l], bankrow[(size_t)l * F_ + f], acc);
        out[((size_t)b * H_ + h) * F_ + f] = acc >= 0.0f ? acc : NEG_SLOPE * acc;
    }
}

// ---------------------------------------------------------------------------
extern "C" void kernel_launch(void* const* d_in, const int* in_sizes, int n_in,
                              void* d_out, int out_size, void* d_ws, size_t ws_size,
                              hipStream_t stream) {
    const float* x     = (const float*)d_in[0];   // [B, E]
    const float* bank  = (const float*)d_in[1];   // [B, L, F]
    const int*   mask  = (const int*)  d_in[2];   // [B, L]
    const float* Query = (const float*)d_in[3];   // [H, D, E]
    const float* Key   = (const float*)d_in[4];   // [H, D, F]
    float* out = (float*)d_out;                   // [B, H, F]

    const int nBank  = B_ * L_ * F_;   // 12,582,912
    const int nKey   = H_ * D_ * F_;   //  1,572,864
    const int nQuery = H_ * D_ * E_;   //  3,145,728
    const int nX     = B_ * E_;        //     98,304

    char* ws = (char*)d_ws;
    size_t off = 0;
    auto take = [&](size_t bytes) {
        size_t cur = off;
        off = (off + bytes + 255) & ~(size_t)255;
        return cur;
    };
    bf16_t* bankbf  = (bf16_t*)(ws + take((size_t)nBank  * sizeof(bf16_t)));
    bf16_t* keybf   = (bf16_t*)(ws + take((size_t)nKey   * sizeof(bf16_t)));
    bf16_t* querybf = (bf16_t*)(ws + take((size_t)nQuery * sizeof(bf16_t)));
    bf16_t* xbf     = (bf16_t*)(ws + take((size_t)nX     * sizeof(bf16_t)));
    float*  qv      = (float*) (ws + take((size_t)H_ * B_ * D_ * sizeof(float)));

    cvt_bf16_kernel<<<2048, 256, 0, stream>>>(bank,  bankbf,  nBank);
    cvt_bf16_kernel<<<1024, 256, 0, stream>>>(Key,   keybf,   nKey);
    cvt_bf16_kernel<<<1024, 256, 0, stream>>>(Query, querybf, nQuery);
    cvt_bf16_kernel<<<256,  256, 0, stream>>>(x,     xbf,     nX);

    q_gemm_kernel<<<H_, 256, 0, stream>>>(xbf, querybf, qv);

    attn_fused_kernel<<<H_ * B_, 256, 0, stream>>>(bankbf, keybf, qv, bank,
                                                   mask, out);
}
